// LieConnection_12661563589239
// MI455X (gfx1250) — matrix operand order repaired and checked by reading, not compiled, not run on hardware
//
#include <hip/hip_runtime.h>
#include <math.h>

// CDNA5 / gfx1250, wave32. bf16 WMMA with f32 accumulate.

typedef __attribute__((ext_vector_type(16))) __bf16 v16bf;
typedef __attribute__((ext_vector_type(8)))  float  v8f;

#define DMODEL 512
#define FIB    64
#define RANK   8
#define NSTEPS 4
#define BT     16      // batch rows per workgroup
#define XS     520     // x_mid LDS row stride in bf16 (512 + 8 pad -> conflict-free b128)
#define SB     72      // 64 + 8 pad stride for 64x64 bf16 LDS tiles

union FragBF { v16bf v; uint4 q[2]; };
union Oct    { uint4 q; __bf16 h[8]; };

static __device__ __forceinline__ v8f wmma_bf16(const FragBF a, const FragBF b, v8f c) {
  // D(f32 16x16) = A(16x32 bf16) * B(32x16 bf16) + C
  return __builtin_amdgcn_wmma_f32_16x16x32_bf16(false, a.v, false, b.v, (short)0, c, false, false);
}

// A fragment: row-major [row][k] buffer. lane<16: K = kb+0..7, kb+16..23 ; lane>=16: +8.
static __device__ __forceinline__ FragBF load_a(const __bf16* base, int stride, int row,
                                                int kbase, int khalf) {
  FragBF f;
  const __bf16* p = base + row * stride + kbase + khalf * 8;
  f.q[0] = *(const uint4*)(p);
  f.q[1] = *(const uint4*)(p + 16);
  return f;
}

// B fragment from a column-major ("B-transposed") buffer laid out [n][k].
// lane<16: col n, K = kb+0..7, kb+16..23 ; lane>=16: +8.
static __device__ __forceinline__ FragBF load_bt(const __bf16* base, int stride, int n,
                                                 int kbase, int khalf) {
  FragBF f;
  const __bf16* p = base + n * stride + kbase + khalf * 8;
  f.q[0] = *(const uint4*)(p);
  f.q[1] = *(const uint4*)(p + 16);
  return f;
}

// ---------------- prelude: weight conversion / transposition ----------------
__global__ __launch_bounds__(256) void lie_prep(
    const float* __restrict__ W1, const float* __restrict__ W2,
    const float* __restrict__ G,
    __bf16* __restrict__ w1b, __bf16* __restrict__ gtb, float* __restrict__ w2t)
{
  const int i = blockIdx.x * 256 + threadIdx.x;
  if (i < DMODEL * DMODEL) w1b[i] = (__bf16)W1[i];                  // [n][k] == W1 row-major
  if (i < RANK * FIB * FIB) {                                       // GT[r][n][k] = G[r][k][n]
    const int r = i >> 12, n = (i >> 6) & 63, k = i & 63;
    gtb[i] = (__bf16)G[(r << 12) + (k << 6) + n];
  }
  if (i < DMODEL * RANK) {                                          // W2T[n][r]
    const int n = i >> 3, r = i & 7;
    w2t[i] = W2[r * DMODEL + n];
  }
}

// ---------------- fused: all 4 integration steps per batch tile ----------------
__global__ __launch_bounds__(256) void lie_fused(
    const float* __restrict__ xsrt, const float* __restrict__ xend,
    const float* __restrict__ b1,   const float* __restrict__ b2,
    const float* __restrict__ scl,
    const __bf16* __restrict__ w1b, const __bf16* __restrict__ gtb,
    const float* __restrict__ w2t,  float* __restrict__ out)
{
  __shared__ __align__(16) __bf16 ls_x[BT * XS];          // x_mid tile, bf16 row-major
  __shared__ __align__(16) float  ls_coef[BT * RANK];     // coefficient accumulator
  __shared__ __align__(16) __bf16 ls_adt[2 * FIB * SB];   // Adt^T  [n][k]  (2 elements/round)
  __shared__ __align__(16) __bf16 ls_trn[2 * FIB * SB];   // transport_old^T [n][k]
  __shared__ __align__(16) __bf16 ls_stg[8 * 16 * SB];    // per-wave A-operand restage strip

  const int tid  = threadIdx.x;
  const int w    = tid >> 5;      // wave 0..7
  const int lane = tid & 31;
  const int col  = lane & 15;
  const int hi   = lane >> 4;
  const int b0   = blockIdx.x * BT;
  const float sc = scl[0];

  const int g  = w >> 2;          // which of 2 concurrent batch elements
  const int ws = w & 3;           // 16-row strip within 64x64
  __bf16* adtg = ls_adt + g * FIB * SB;
  __bf16* trng = ls_trn + g * FIB * SB;
  __bf16* stg  = ls_stg + w * 16 * SB;

  #pragma unroll 1
  for (int step = 0; step < NSTEPS; ++step) {
    const float t = ((float)step + 0.5f) * (1.0f / NSTEPS);

    // ---- x_mid tile -> LDS (bf16), coalesced float4 ----
    #pragma unroll 1
    for (int i = tid; i < BT * DMODEL / 4; i += 256) {
      const int row = i >> 7;
      const int d4  = (i & 127) << 2;
      const float4 a = *(const float4*)(xsrt + (size_t)(b0 + row) * DMODEL + d4);
      const float4 c = *(const float4*)(xend + (size_t)(b0 + row) * DMODEL + d4);
      __bf16* p = ls_x + row * XS + d4;
      p[0] = (__bf16)((1.0f - t) * a.x + t * c.x);
      p[1] = (__bf16)((1.0f - t) * a.y + t * c.y);
      p[2] = (__bf16)((1.0f - t) * a.z + t * c.z);
      p[3] = (__bf16)((1.0f - t) * a.w + t * c.w);
    }
    if (tid < BT * RANK) ls_coef[tid] = 0.0f;
    __syncthreads();

    // ---- GEMM1: h = GELU(x_mid @ W1^T + b1), fused 8-wide W2 projection ----
    {
      v8f acc[4] = {};
      #pragma unroll 1          // keep only one iteration's fragments live (no spills)
      for (int kc = 0; kc < DMODEL / 32; ++kc) {
        const FragBF A = load_a(ls_x, XS, col, kc * 32, hi);   // 16 batch rows
        #pragma unroll
        for (int tt = 0; tt < 4; ++tt) {
          const int n = w * 64 + tt * 16 + col;                // 64 neurons per wave
          const FragBF B = load_bt(w1b, DMODEL, n, kc * 32, hi);
          acc[tt] = wmma_bf16(A, B, acc[tt]);
        }
      }
      float part[8][RANK];
      #pragma unroll
      for (int v = 0; v < 8; ++v)
        #pragma unroll
        for (int r = 0; r < RANK; ++r) part[v][r] = 0.0f;
      #pragma unroll 1
      for (int tt = 0; tt < 4; ++tt) {
        const int n = w * 64 + tt * 16 + col;
        const float bias = b1[n];
        const float4 w2lo = *(const float4*)(w2t + n * RANK);
        const float4 w2hi = *(const float4*)(w2t + n * RANK + 4);
        #pragma unroll
        for (int v = 0; v < 8; ++v) {
          const float h = acc[tt][v] + bias;
          const float gl = 0.5f * h * (1.0f + erff(h * 0.70710678118f)); // exact GELU
          part[v][0] += gl * w2lo.x; part[v][1] += gl * w2lo.y;
          part[v][2] += gl * w2lo.z; part[v][3] += gl * w2lo.w;
          part[v][4] += gl * w2hi.x; part[v][5] += gl * w2hi.y;
          part[v][6] += gl * w2hi.z; part[v][7] += gl * w2hi.w;
        }
      }
      // butterfly reduce across each 16-lane half (same batch rows, different neurons)
      #pragma unroll 1
      for (int v = 0; v < 8; ++v)
        #pragma unroll
        for (int r = 0; r < RANK; ++r) {
          float x = part[v][r];
          x += __shfl_xor(x, 1, 32);
          x += __shfl_xor(x, 2, 32);
          x += __shfl_xor(x, 4, 32);
          x += __shfl_xor(x, 8, 32);
          if (col == 0) atomicAdd(&ls_coef[(v + hi * 8) * RANK + r], x);
        }
    }
    __syncthreads();
    if (tid < BT * RANK) {
      const int r = tid & 7;
      ls_coef[tid] = tanhf(ls_coef[tid] + b2[r]) * sc * (1.0f / NSTEPS); // coeff * scale * dt
    }
    __syncthreads();

    // ---- exp-series chain + transport update: 2 batch elements x 4 waves per round ----
    #pragma unroll 1
    for (int rnd = 0; rnd < BT / 2; ++rnd) {
      const int e = rnd * 2 + g;
      const size_t ob = (size_t)(b0 + e) * (FIB * FIB);

      { // build Adt^T (and transport_old^T) for this element; 128 threads per element
        float cf[RANK];
        #pragma unroll
        for (int r = 0; r < RANK; ++r) cf[r] = ls_coef[e * RANK + r];
        const int lt = ws * 32 + lane;
        #pragma unroll 1
        for (int idx = lt; idx < FIB * FIB; idx += 128) {
          const int n = idx >> 6, k = idx & 63;
          float a = 0.0f;
          #pragma unroll
          for (int r = 0; r < RANK; ++r)
            a += cf[r] * (float)gtb[(r << 12) + idx];     // GT[r][n][k] = G[r][k][n]
          adtg[n * SB + k] = (__bf16)a;
          if (step > 0) trng[n * SB + k] = (__bf16)out[ob + k * FIB + n];
        }
      }
      __syncthreads();

      // S1 = Adt strip (f32), dT = I + Adt strip
      v8f term[4], dT[4];
      const int mb = ws * 16 + hi * 8;
      #pragma unroll
      for (int tt = 0; tt < 4; ++tt) {
        const int n = tt * 16 + col;
        Oct o; o.q = *(const uint4*)(adtg + n * SB + mb);  // Adt^T[n][mb..mb+7]
        #pragma unroll
        for (int v = 0; v < 8; ++v) {
          const float a = (float)o.h[v];
          term[tt][v] = a;
          dT[tt][v]   = a + ((mb + v) == n ? 1.0f : 0.0f);
        }
      }
      #pragma unroll
      for (int tt = 0; tt < 4; ++tt)
        #pragma unroll
        for (int v = 0; v < 8; ++v)
          stg[(v + hi * 8) * SB + tt * 16 + col] = (__bf16)term[tt][v];
      __builtin_amdgcn_wave_barrier();

      // S_k = S_{k-1} @ Adt * (1/k)  -> Adt^k / k!   (wave-private, no block barriers)
      #pragma unroll 1
      for (int k = 2; k <= 5; ++k) {
        const float rk = 1.0f / (float)k;
        v8f nt[4] = {};
        #pragma unroll
        for (int kc = 0; kc < 2; ++kc) {
          const FragBF A = load_a(stg, SB, col, kc * 32, hi);
          #pragma unroll
          for (int tt = 0; tt < 4; ++tt) {
            const FragBF B = load_bt(adtg, SB, tt * 16 + col, kc * 32, hi);
            nt[tt] = wmma_bf16(A, B, nt[tt]);
          }
        }
        #pragma unroll
        for (int tt = 0; tt < 4; ++tt)
          #pragma unroll
          for (int v = 0; v < 8; ++v) {
            const float s = nt[tt][v] * rk;
            term[tt][v] = s;
            dT[tt][v]  += s;
          }
        if (k < 5) {
          #pragma unroll
          for (int tt = 0; tt < 4; ++tt)
            #pragma unroll
            for (int v = 0; v < 8; ++v)
              stg[(v + hi * 8) * SB + tt * 16 + col] = (__bf16)term[tt][v];
          __builtin_amdgcn_wave_barrier();
        }
      }

      // transport_new = dT @ transport_old (step 0: transport_old = I)
      v8f tr[4] = {};
      if (step == 0) {
        #pragma unroll
        for (int tt = 0; tt < 4; ++tt) tr[tt] = dT[tt];
      } else {
        #pragma unroll
        for (int tt = 0; tt < 4; ++tt)
          #pragma unroll
          for (int v = 0; v < 8; ++v)
            stg[(v + hi * 8) * SB + tt * 16 + col] = (__bf16)dT[tt][v];
        __builtin_amdgcn_wave_barrier();
        #pragma unroll
        for (int kc = 0; kc < 2; ++kc) {
          const FragBF A = load_a(stg, SB, col, kc * 32, hi);
          #pragma unroll
          for (int tt = 0; tt < 4; ++tt) {
            const FragBF B = load_bt(trng, SB, tt * 16 + col, kc * 32, hi);
            tr[tt] = wmma_bf16(A, B, tr[tt]);
          }
        }
      }
      #pragma unroll
      for (int tt = 0; tt < 4; ++tt) {
        const int n = tt * 16 + col;
        #pragma unroll
        for (int v = 0; v < 8; ++v)
          out[ob + (size_t)(mb + v) * FIB + n] = tr[tt][v];   // f32, lanes 0-15 coalesce 64B
      }
      __syncthreads();
    }
    __threadfence();     // transport written this step is re-read next step
    __syncthreads();
  }
}

extern "C" void kernel_launch(void* const* d_in, const int* in_sizes, int n_in,
                              void* d_out, int out_size, void* d_ws, size_t ws_size,
                              hipStream_t stream) {
  const float* xs = (const float*)d_in[0];
  const float* xe = (const float*)d_in[1];
  const float* W1 = (const float*)d_in[2];
  const float* b1 = (const float*)d_in[3];
  const float* W2 = (const float*)d_in[4];
  const float* b2 = (const float*)d_in[5];
  const float* G  = (const float*)d_in[6];
  const float* sc = (const float*)d_in[7];
  float* out = (float*)d_out;

  char* ws = (char*)d_ws;
  __bf16* w1b = (__bf16*)(ws);                                     // 512 KB
  __bf16* gtb = (__bf16*)(ws + DMODEL * DMODEL * 2);               //  64 KB
  float*  w2t = (float*)(ws + DMODEL * DMODEL * 2 + RANK * FIB * FIB * 2); // 16 KB

  lie_prep<<<(DMODEL * DMODEL) / 256, 256, 0, stream>>>(W1, W2, G, w1b, gtb, w2t);

  const int nblocks = in_sizes[0] / DMODEL / BT;                   // 16384 / 16 = 1024
  lie_fused<<<nblocks, 256, 0, stream>>>(xs, xe, b1, b2, sc, w1b, gtb, w2t, out);
}
